// SingleLSTMQuantization_58291296141791
// MI455X (gfx1250) — matrix operand order repaired and checked by reading, not compiled, not run on hardware
//
#include <hip/hip_runtime.h>
#include <hip/hip_bf16.h>
#include <math.h>

// CDNA5 wave32 WMMA types
typedef __attribute__((ext_vector_type(16))) _Float16 v16h;
typedef __attribute__((ext_vector_type(8)))  _Float16 v8h;
typedef __attribute__((ext_vector_type(8)))  float    v8f;

#define S_TOT   256
#define B_TOT   4096
#define H_DIM   120
#define G4      480
#define NSTEP   36        // only t = 0..35 feed the output (hs[35])
#define MID     10

#define M_TILE  32        // batch rows per workgroup (2 WMMA M-tiles)
#define KPAD    128       // H padded to 4 WMMA k-steps of 32
#define NT_TOT  32        // 4 gates * 8 n-tiles (each gate padded 120 -> 128)
#define THREADS 256       // 8 waves; each wave owns 4 n-tiles x both M-tiles

// ---- LDS layout (bytes): no B staging needed (B lives in registers) ----
#define OFF_G    0
#define SZ_G     (M_TILE * 512 * 4)       // 65536  (gate pre-activations, f32)
#define OFF_H    (OFF_G + SZ_G)
#define SZ_H     (M_TILE * KPAD * 2)      // 8192   (h state, f16)
#define OFF_C    (OFF_H + SZ_H)
#define SZ_C     (M_TILE * KPAD * 4)      // 16384  (c state, f32)
#define OFF_BIAS (OFF_C + SZ_C)
#define SZ_BIAS  (G4 * 4)                 // 1920   (b_ih + b_hh)
#define OFF_WIH  (OFF_BIAS + SZ_BIAS)
#define SZ_WIH   (G4 * 4)                 // 1920   (w_ih, IN==1)
#define LDS_TOTAL (OFF_WIH + SZ_WIH)      // 93952 bytes

__device__ __forceinline__ float sig_f(float x) {
    return 1.0f / (1.0f + __expf(-x));
}

__global__ __launch_bounds__(THREADS, 1)
void lstm36_wmma_f16(const float* __restrict__ x,
                     const float* __restrict__ w_ih,
                     const float* __restrict__ w_hh,
                     const float* __restrict__ b_ih,
                     const float* __restrict__ b_hh,
                     const float* __restrict__ l1_w,
                     const float* __restrict__ l1_b,
                     const float* __restrict__ l2_w,
                     const float* __restrict__ l2_b,
                     float* __restrict__ out)
{
    extern __shared__ char smem[];
    float*    ldsG    = (float*)   (smem + OFF_G);
    _Float16* ldsH    = (_Float16*)(smem + OFF_H);
    float*    ldsC    = (float*)   (smem + OFF_C);
    float*    ldsBias = (float*)   (smem + OFF_BIAS);
    float*    ldsWih  = (float*)   (smem + OFF_WIH);

    const int tid    = threadIdx.x;
    const int lane   = tid & 31;
    const int wv     = tid >> 5;                // 0..7
    const int wgBase = blockIdx.x * M_TILE;

    // ---------------- init ----------------
    if (tid < G4) {
        ldsBias[tid] = b_ih[tid] + b_hh[tid];
        ldsWih[tid]  = w_ih[tid];               // IN == 1
    }
    for (int e = tid; e < M_TILE * KPAD; e += THREADS) {
        ldsH[e] = (_Float16)0.0f;               // h0 = 0 (pad cols stay 0 forever)
        ldsC[e] = 0.0f;                         // c0 = 0
    }

    // Gather this wave's loop-invariant B fragments of w_hh^T straight from
    // global into registers (held live across all 36 steps).
    // Fragment (ntile, ks): lane l -> N col = ntile*16 + l%16,
    //   K = ks*32 + (l<16 ? {0..7,16..23} : {8..15,24..31}); w_hh row = gate*120 + col.
    const int ntbase = wv * 4;                  // 4 n-tiles per wave
    v16h Bf[16];                                // [nt*4 + ks], 128 VGPRs
    {
        const int nloc  = lane & 15;
        const int khalf = (lane & 16) ? 8 : 0;
        #pragma unroll
        for (int nt = 0; nt < 4; ++nt) {
            int ntile = ntbase + nt;
            int gate  = ntile >> 3;
            int col   = (ntile & 7) * 16 + nloc;
            bool colok = (col < H_DIM);
            const float* srow = w_hh + (size_t)(gate * H_DIM + (colok ? col : 0)) * H_DIM;
            #pragma unroll
            for (int ks = 0; ks < 4; ++ks) {
                int kbase = ks * 32 + khalf;
                v16h f;
                #pragma unroll
                for (int kk = 0; kk < 16; ++kk) {
                    int k = kbase + (kk < 8 ? kk : kk + 8);
                    f[kk] = (colok && k < H_DIM) ? (_Float16)srow[k] : (_Float16)0.0f;
                }
                Bf[nt * 4 + ks] = f;
            }
        }
    }
    __syncthreads();

    // ---------------- recurrence: 36 steps ----------------
    const int arow0 = lane & 15;                // A-matrix M row within tile
    const int aksel = (lane & 16) ? 8 : 0;      // A per-lane K half-select
    const int nbase = lane & 15;
    const int msub  = (lane & 16) ? 8 : 0;

    #pragma unroll 1
    for (int t = 0; t < NSTEP; ++t) {
        v8f acc[8] = {};                        // [mt*4 + nt]

        #pragma unroll
        for (int mt = 0; mt < 2; ++mt) {
            #pragma unroll
            for (int ks = 0; ks < 4; ++ks) {
                // A: lane<16 holds K {0..7,16..23}+32ks, lane>=16 holds {8..15,24..31}+32ks
                const _Float16* hrow = ldsH + (mt * 16 + arow0) * KPAD + ks * 32 + aksel;
                v8h p0 = *(const v8h*)(hrow);
                v8h p1 = *(const v8h*)(hrow + 16);
                v16h a = __builtin_shufflevector(p0, p1,
                            0,1,2,3,4,5,6,7,8,9,10,11,12,13,14,15);
                #pragma unroll
                for (int nt = 0; nt < 4; ++nt) {
                    acc[mt * 4 + nt] = __builtin_amdgcn_wmma_f32_16x16x32_f16(
                        false, a, false, Bf[nt * 4 + ks], (short)0,
                        acc[mt * 4 + nt], false, false);
                }
            }
        }

        // D layout: lane l, vgpr v -> M = v + 8*(l>=16), N = l%16
        #pragma unroll
        for (int mt = 0; mt < 2; ++mt) {
            int mrow0 = mt * 16 + msub;
            #pragma unroll
            for (int nt = 0; nt < 4; ++nt) {
                int n = (ntbase + nt) * 16 + nbase;
                #pragma unroll
                for (int v = 0; v < 8; ++v)
                    ldsG[(mrow0 + v) * 512 + n] = acc[mt * 4 + nt][v];
            }
        }
        __syncthreads();

        // elementwise gate fusion: 32*120 = 3840 = 15 * 256 (no divergence)
        #pragma unroll 1
        for (int it = 0; it < (M_TILE * H_DIM) / THREADS; ++it) {
            int e   = tid + it * THREADS;
            int m   = e / H_DIM;
            int col = e - m * H_DIM;
            float xv = x[(size_t)t * B_TOT + wgBase + m];   // IN == 1
            const float* gr = ldsG + m * 512;
            float gi = gr[      col] + xv * ldsWih[      col] + ldsBias[      col];
            float gf = gr[128 + col] + xv * ldsWih[120 + col] + ldsBias[120 + col];
            float gc = gr[256 + col] + xv * ldsWih[240 + col] + ldsBias[240 + col];
            float go = gr[384 + col] + xv * ldsWih[360 + col] + ldsBias[360 + col];
            float i_ = sig_f(gi);
            float f_ = sig_f(gf);
            float g_ = tanhf(gc);
            float o_ = sig_f(go);
            float c_ = f_ * ldsC[m * KPAD + col] + i_ * g_;
            float h_ = o_ * tanhf(c_);
            ldsC[m * KPAD + col] = c_;
            ldsH[m * KPAD + col] = (_Float16)h_;
        }
        __syncthreads();
    }

    // ---------------- head: relu(h @ l1^T + b1) @ l2^T + b2 ----------------
    if (tid < M_TILE) {
        int m = tid;
        float y = l2_b[0];
        #pragma unroll 1
        for (int j = 0; j < MID; ++j) {
            float s = l1_b[j];
            const float* wr = l1_w + j * H_DIM;
            #pragma unroll 4
            for (int k = 0; k < H_DIM; ++k)
                s += wr[k] * (float)ldsH[m * KPAD + k];
            y += l2_w[j] * fmaxf(s, 0.0f);
        }
        out[wgBase + m] = y;
    }
}

extern "C" void kernel_launch(void* const* d_in, const int* in_sizes, int n_in,
                              void* d_out, int out_size, void* d_ws, size_t ws_size,
                              hipStream_t stream) {
    (void)in_sizes; (void)n_in; (void)out_size; (void)d_ws; (void)ws_size;
    const float* x    = (const float*)d_in[0];
    const float* w_ih = (const float*)d_in[1];
    const float* w_hh = (const float*)d_in[2];
    const float* b_ih = (const float*)d_in[3];
    const float* b_hh = (const float*)d_in[4];
    const float* l1_w = (const float*)d_in[5];
    const float* l1_b = (const float*)d_in[6];
    const float* l2_w = (const float*)d_in[7];
    const float* l2_b = (const float*)d_in[8];
    float* out = (float*)d_out;

    dim3 grid(B_TOT / M_TILE);   // 128 workgroups, one per 32 batch rows
    dim3 block(THREADS);         // 8 wave32 waves
    lstm36_wmma_f16<<<grid, block, LDS_TOTAL, stream>>>(
        x, w_ih, w_hh, b_ih, b_hh, l1_w, l1_b, l2_w, l2_b, out);
}
